// FirenetSparseObjectDet_70686571758044
// MI455X (gfx1250) — compile-verified, matrix operand order checked by reading, not run on hardware
//
#include <hip/hip_runtime.h>
#include <hip/hip_bf16.h>

typedef __attribute__((ext_vector_type(16))) _Float16 v16h;
typedef __attribute__((ext_vector_type(8)))  float    v8f;
typedef __attribute__((ext_vector_type(4)))  unsigned vu4;
typedef __attribute__((ext_vector_type(8)))  int      vi8;
typedef __attribute__((ext_vector_type(4)))  int      vi4;

union HFrag { v16h h; unsigned u[8]; };

// Read a 16-bit A/B WMMA fragment from an LDS tile stored as [16 rows][32 k] f16
// (viewed as [16][16] u32 pairs). Per ISA 7.12.2: lane holds row = lane&15,
// half = lane>>4; element pair v covers K = (v<4?0:16) + half*8 + (v&3)*2.
__device__ __forceinline__ v16h frag_from_lds(const unsigned* lds32, int row, int half) {
  HFrag f;
#pragma unroll
  for (int v = 0; v < 8; ++v) {
    const int p = (v < 4 ? 0 : 8) + half * 4 + (v & 3);
    f.u[v] = lds32[row * 16 + p];
  }
  return f.h;
}

// Load 16 contiguous floats with 4x b128 loads (p must be 16B aligned & valid).
__device__ __forceinline__ void load16(float* vv, const float* p) {
  const float4* q = (const float4*)p;
  const float4 a = q[0], b = q[1], c = q[2], d = q[3];
  vv[0]  = a.x; vv[1]  = a.y; vv[2]  = a.z; vv[3]  = a.w;
  vv[4]  = b.x; vv[5]  = b.y; vv[6]  = b.z; vv[7]  = b.w;
  vv[8]  = c.x; vv[9]  = c.y; vv[10] = c.z; vv[11] = c.w;
  vv[12] = d.x; vv[13] = d.y; vv[14] = d.z; vv[15] = d.w;
}

// Convert 16 floats -> 8 packed half2 dwords in LDS (merges into ds_store_b128s).
__device__ __forceinline__ void cvt_store16(unsigned* d32, const float* vv) {
  union { _Float16 h[2]; unsigned u; } pk;
#pragma unroll
  for (int p = 0; p < 8; ++p) {
    pk.h[0] = (_Float16)vv[2 * p];
    pk.h[1] = (_Float16)vv[2 * p + 1];
    d32[p] = pk.u;
  }
}

// Issue a TDM async DMA of a [rows x 32] f32 tile (row stride = strideElems)
// from global memory into LDS at byte offset ldsOff. D# built per ISA ch.8:
// group0 = {count=1, lds_addr, global_addr[56:0], type=2}; group1 = dims/tile.
__device__ __forceinline__ void tdm_load_tile_f32(const float* gsrc, unsigned ldsOff,
                                                  int strideElems, int rows) {
  const unsigned long long ga = (unsigned long long)(size_t)gsrc;
  vu4 g0 = {};
  g0[0] = 1u;                                         // count=1, user mode
  g0[1] = ldsOff;                                     // lds_addr (bytes)
  g0[2] = (unsigned)(ga & 0xffffffffu);               // global_addr[31:0]
  g0[3] = (unsigned)((ga >> 32) & 0x01ffffffu)        // global_addr[56:32]
        | 0x80000000u;                                // type = 2 ("image")
  vi8 g1 = {};
  g1[0] = 0x00020000;                                 // data_size = 2 -> 4 bytes
  g1[1] = (int)(((unsigned)strideElems & 0xffffu) << 16);         // tensor_dim0[15:0]
  g1[2] = (int)(((unsigned)strideElems >> 16) | ((unsigned)rows << 16)); // dim0 hi | dim1 lo
  g1[3] = (int)(32u << 16);                           // tile_dim0 = 32
  g1[4] = rows;                                       // tile_dim1 = rows
  g1[5] = strideElems;                                // tensor_dim0_stride
  const vi4 z4 = {};
#if __clang_major__ >= 23
  const vi8 z8 = {};
  __builtin_amdgcn_tensor_load_to_lds(g0, g1, z4, z4, z8, 0);
#else
  __builtin_amdgcn_tensor_load_to_lds(g0, g1, z4, z4, 0);
#endif
}

// ---------------------------------------------------------------------------
// Implicit-GEMM submanifold 3x3 conv, FAST variant: requires C0 % 32 == 0.
// One wave computes 16 sites x (NT*16) out channels. All staging loads are
// unconditional b128s from clamped-valid addresses; zeroing is done by
// multiplying with a 0/1 float mask (keeps loads branchless, no exec dances).
// C0 = real input channels; CiW = weight input-channel stride (CiW > C0
// models concat(x, h) with h == 0 -> those K chunks are skipped entirely).
// ---------------------------------------------------------------------------
template <int NT>
__global__ __launch_bounds__(32)
void fnet_conv3x3_wmma(const float* __restrict__ src, int C0, int CiW,
                       const float* __restrict__ Wt,     // [3][3][CiW][Co]
                       const float* __restrict__ maskL,  // [B*H*W] 0/1
                       float* __restrict__ dst,          // [B*H*W][Co]
                       int Bn, int H, int Wd, int Co)
{
  __shared__ unsigned ldsA32[16 * 16];
  __shared__ unsigned ldsB32[NT * 16 * 16];
  _Float16* ldsB = reinterpret_cast<_Float16*>(ldsB32);

  const int lane   = threadIdx.x;
  const int HW     = H * Wd;
  const int S      = Bn * HW;
  const int tileM  = blockIdx.x;
  const int tileN0 = blockIdx.y * NT;
  const int mrow   = lane & 15;
  const int chalf  = lane >> 4;

  const int siteA = tileM * 16 + mrow;
  const bool rowValid = siteA < S;
  const int siteC = rowValid ? siteA : 0;
  const int bA = siteC / HW;
  const int rem = siteC - bA * HW;
  const int yA = rem / Wd;
  const int xA = rem - yA * Wd;

  const v8f vzero = {};
  v8f acc[NT];
#pragma unroll
  for (int t = 0; t < NT; ++t) acc[t] = vzero;

#pragma unroll
  for (int ky = 0; ky < 3; ++ky) {
#pragma unroll
    for (int kx = 0; kx < 3; ++kx) {
      const int iy = yA + ky - 1;
      const int ix = xA + kx - 1;
      const bool inb = rowValid && (iy >= 0) && (iy < H) && (ix >= 0) && (ix < Wd);
      const float inbF = inb ? 1.0f : 0.0f;
      const float* abase = inb ? (src + (size_t)((bA * H + iy) * Wd + ix) * C0) : src;
      const int kwbase = (ky * 3 + kx) * CiW;

      for (int c0 = 0; c0 < C0; c0 += 32) {
        // ---- stage A chunk (16 sites x 32 channels): 4x b128 + mask-mul ----
        float vv[16];
        load16(vv, abase + c0 + chalf * 16);
#pragma unroll
        for (int j = 0; j < 16; ++j) vv[j] *= inbF;
        cvt_store16(&ldsA32[mrow * 16 + chalf * 8], vv);

        // ---- stage B chunk transposed; k-row always valid here:
        //      c0 + lane < C0 <= CiW since C0 % 32 == 0 ----
        {
          const float* wr = Wt + (size_t)(kwbase + c0 + lane) * Co + tileN0 * 16;
          float wvv[NT * 16];
#pragma unroll
          for (int t = 0; t < NT; ++t) load16(&wvv[t * 16], wr + t * 16);
#pragma unroll
          for (int t = 0; t < NT; ++t)
#pragma unroll
            for (int j = 0; j < 16; ++j)
              ldsB[t * 512 + j * 32 + lane] = (_Float16)wvv[t * 16 + j];
        }

        const v16h a = frag_from_lds(ldsA32, mrow, chalf);
#pragma unroll
        for (int t = 0; t < NT; ++t) {
          const v16h b = frag_from_lds(ldsB32 + t * 256, mrow, chalf);
          acc[t] = __builtin_amdgcn_wmma_f32_16x16x32_f16(false, a, false, b,
                                                          (short)0, acc[t], false, false);
        }
      }
    }
  }

  const int n     = lane & 15;
  const int mhalf = lane >> 4;
#pragma unroll
  for (int r = 0; r < 8; ++r) {
    const int site = tileM * 16 + mhalf * 8 + r;
    const bool ok = site < S;
    const float mv = maskL[ok ? site : 0];
#pragma unroll
    for (int t = 0; t < NT; ++t) {
      const int col = (tileN0 + t) * 16 + n;
      if (ok && col < Co)
        dst[(size_t)site * Co + col] = (mv > 0.0f) ? acc[t][r] : 0.0f;
    }
  }
}

// ---------------------------------------------------------------------------
// Small-channel conv (C0T in {2,16}, C0 == CiW): the weight k-index
// (ky*3+kx)*C0 + ci is CONTIGUOUS in k, so multiple taps pack into each
// 32-wide K chunk (16 taps at C0T=2, 2 taps at C0T=16). Fully vectorized,
// branchless staging; K = 9*C0T padded to chunks of 32.
// ---------------------------------------------------------------------------
template <int NT, int C0T>
__global__ __launch_bounds__(32)
void fnet_conv3x3_smallc_wmma(const float* __restrict__ src,
                              const float* __restrict__ Wt,    // [9*C0T][Co]
                              const float* __restrict__ maskL,
                              float* __restrict__ dst,
                              int Bn, int H, int Wd, int Co)
{
  constexpr int K   = 9 * C0T;
  constexpr int KCH = (K + 31) / 32;   // 1 for C0T=2, 5 for C0T=16
  constexpr int G   = 16 / C0T;        // tap-groups per 16-element lane slice

  __shared__ unsigned ldsA32[16 * 16];
  __shared__ unsigned ldsB32[NT * 16 * 16];
  _Float16* ldsB = reinterpret_cast<_Float16*>(ldsB32);

  const int lane   = threadIdx.x;
  const int HW     = H * Wd;
  const int S      = Bn * HW;
  const int tileM  = blockIdx.x;
  const int tileN0 = blockIdx.y * NT;
  const int mrow   = lane & 15;
  const int chalf  = lane >> 4;

  const int siteA = tileM * 16 + mrow;
  const bool rowValid = siteA < S;
  const int siteC = rowValid ? siteA : 0;
  const int bA = siteC / HW;
  const int rem = siteC - bA * HW;
  const int yA = rem / Wd;
  const int xA = rem - yA * Wd;

  const v8f vzero = {};
  v8f acc[NT];
#pragma unroll
  for (int t = 0; t < NT; ++t) acc[t] = vzero;

  for (int ch = 0; ch < KCH; ++ch) {
    const int kbase = ch * 32;
    float vv[16];
#pragma unroll
    for (int g = 0; g < G; ++g) {
      const int tap  = (kbase + chalf * 16) / C0T + g;   // C0T pow2 -> shift
      const int tapc = (tap < 9) ? tap : 8;
      const int ky = tapc / 3, kx = tapc - 3 * ky;
      const int iy = yA + ky - 1;
      const int ix = xA + kx - 1;
      const bool ok = rowValid && (tap < 9) &&
                      (iy >= 0) && (iy < H) && (ix >= 0) && (ix < Wd);
      const float okF = ok ? 1.0f : 0.0f;
      const float* bp = ok ? (src + (size_t)((bA * H + iy) * Wd + ix) * C0T) : src;
      if constexpr (C0T == 2) {
        const float2 q = *(const float2*)bp;             // b64 load
        vv[2 * g + 0] = q.x * okF;
        vv[2 * g + 1] = q.y * okF;
      } else {
        float tmp[16];
        load16(tmp, bp);                                  // 4x b128
#pragma unroll
        for (int j = 0; j < 16; ++j) vv[g * 16 + j] = tmp[j] * okF;
      }
    }
    cvt_store16(&ldsA32[mrow * 16 + chalf * 8], vv);

    // ---- stage B: lane owns contiguous weight k-row kbase+lane ----
    {
      const int k = kbase + lane;
      const bool kv = k < K;
      const float kvF = kv ? 1.0f : 0.0f;
      const float* wr = Wt + (size_t)(kv ? k : 0) * Co + tileN0 * 16;
      float wvv[NT * 16];
#pragma unroll
      for (int t = 0; t < NT; ++t) load16(&wvv[t * 16], wr + t * 16);
#pragma unroll
      for (int i = 0; i < NT * 16; ++i) wvv[i] *= kvF;
#pragma unroll
      for (int t = 0; t < NT; ++t)
#pragma unroll
        for (int j = 0; j < 16; ++j)
          ldsB[t * 512 + j * 32 + lane] = (_Float16)wvv[t * 16 + j];
    }

    const v16h a = frag_from_lds(ldsA32, mrow, chalf);
#pragma unroll
    for (int t = 0; t < NT; ++t) {
      const v16h b = frag_from_lds(ldsB32 + t * 256, mrow, chalf);
      acc[t] = __builtin_amdgcn_wmma_f32_16x16x32_f16(false, a, false, b,
                                                      (short)0, acc[t], false, false);
    }
  }

  const int n     = lane & 15;
  const int mhalf = lane >> 4;
#pragma unroll
  for (int r = 0; r < 8; ++r) {
    const int site = tileM * 16 + mhalf * 8 + r;
    const bool ok = site < S;
    const float mv = maskL[ok ? site : 0];
#pragma unroll
    for (int t = 0; t < NT; ++t) {
      const int col = (tileN0 + t) * 16 + n;
      if (ok && col < Co)
        dst[(size_t)site * Co + col] = (mv > 0.0f) ? acc[t][r] : 0.0f;
    }
  }
}

// ---------------------------------------------------------------------------
// Dense GEMM: C[M,N] = relu?(A[M,K] * B[K,N] + bias). 16 x (NT*16) per wave.
// A tiles are DMA'd into LDS by the Tensor Data Mover (tensor_load_to_lds +
// s_wait_tensorcnt); B staged with branchless b128 loads. Requires K % 32 == 0
// and M % 16 == 0 (true for all call sites here: M=32).
// ---------------------------------------------------------------------------
template <int NT>
__global__ __launch_bounds__(32)
void fnet_gemm_wmma(const float* __restrict__ A, const float* __restrict__ Bm,
                    const float* __restrict__ bias, float* __restrict__ Cout,
                    int M, int N, int K, int doRelu)
{
  __shared__ unsigned ldsA32[16 * 16];
  __shared__ unsigned ldsB32[NT * 16 * 16];
  __shared__ float    stageA[16 * 32];     // TDM destination (f32 A tile)
  _Float16* ldsB = reinterpret_cast<_Float16*>(ldsB32);

  const int lane   = threadIdx.x;
  const int tileM  = blockIdx.x;
  const int tileN0 = blockIdx.y * NT;
  const int mrow   = lane & 15;
  const int chalf  = lane >> 4;
  const bool nFull = (tileN0 + NT) * 16 <= N;   // wave-uniform
  const unsigned stageOff = (unsigned)(size_t)(void*)stageA;  // LDS byte offset

  const v8f vzero = {};
  v8f acc[NT];
#pragma unroll
  for (int t = 0; t < NT; ++t) acc[t] = vzero;

  for (int k0 = 0; k0 < K; k0 += 32) {
    // ---- TDM: async-DMA the 16x32 f32 A tile into LDS, then convert ----
    {
      tdm_load_tile_f32(A + (size_t)tileM * 16 * K + k0, stageOff, K, 16);
      __builtin_amdgcn_s_wait_tensorcnt(0);
      float vv[16];
      load16(vv, &stageA[mrow * 32 + chalf * 16]);
      cvt_store16(&ldsA32[mrow * 16 + chalf * 8], vv);
    }
    // ---- stage B chunk transposed ----
    {
      const int kk = k0 + lane;                 // kk < K since K % 32 == 0
      const float* br = Bm + (size_t)kk * N;
      float wvv[NT * 16];
      if (nFull) {
#pragma unroll
        for (int t = 0; t < NT; ++t) load16(&wvv[t * 16], br + tileN0 * 16 + t * 16);
      } else {
#pragma unroll
        for (int t = 0; t < NT; ++t)
#pragma unroll
          for (int j = 0; j < 16; ++j) {
            const int col = (tileN0 + t) * 16 + j;      // wave-uniform
            const float cF = (col < N) ? 1.0f : 0.0f;
            wvv[t * 16 + j] = br[(col < N) ? col : 0] * cF;
          }
      }
#pragma unroll
      for (int t = 0; t < NT; ++t)
#pragma unroll
        for (int j = 0; j < 16; ++j)
          ldsB[t * 512 + j * 32 + lane] = (_Float16)wvv[t * 16 + j];
    }
    const v16h a = frag_from_lds(ldsA32, mrow, chalf);
#pragma unroll
    for (int t = 0; t < NT; ++t) {
      const v16h b = frag_from_lds(ldsB32 + t * 256, mrow, chalf);
      acc[t] = __builtin_amdgcn_wmma_f32_16x16x32_f16(false, a, false, b,
                                                      (short)0, acc[t], false, false);
    }
  }

  const int n     = lane & 15;
  const int mhalf = lane >> 4;
#pragma unroll
  for (int r = 0; r < 8; ++r) {
    const int rw = tileM * 16 + mhalf * 8 + r;
#pragma unroll
    for (int t = 0; t < NT; ++t) {
      const int col = (tileN0 + t) * 16 + n;
      if (rw < M && col < N) {
        float v = acc[t][r] + (bias ? bias[col] : 0.0f);
        if (doRelu) v = fmaxf(v, 0.0f);
        Cout[(size_t)rw * N + col] = v;
      }
    }
  }
}

// ---------------------------------------------------------------------------
// Elementwise / reduction helpers (bandwidth-bound; plain VALU kernels).
// ---------------------------------------------------------------------------
__global__ void fnet_zero_k(float* p, long long n) {
  const long long stride = (long long)gridDim.x * blockDim.x;
  for (long long i = blockIdx.x * (long long)blockDim.x + threadIdx.x; i < n; i += stride)
    p[i] = 0.0f;
}

__global__ void fnet_mask_cvt_k(const unsigned char* m, float* out, long long n) {
  const long long stride = (long long)gridDim.x * blockDim.x;
  for (long long i = blockIdx.x * (long long)blockDim.x + threadIdx.x; i < n; i += stride)
    out[i] = m[i] ? 1.0f : 0.0f;
}

__global__ void fnet_bnstats_k(const float* __restrict__ src, float* sum, float* sumsq,
                               long long S, int C)
{
  const int c = blockIdx.x;
  float s = 0.0f, ss = 0.0f;
  const long long stride = (long long)gridDim.y * blockDim.x;
  for (long long i = blockIdx.y * (long long)blockDim.x + threadIdx.x; i < S; i += stride) {
    const float v = src[i * C + c];
    s += v; ss += v * v;
  }
  __shared__ float r0[256], r1[256];
  r0[threadIdx.x] = s; r1[threadIdx.x] = ss;
  __syncthreads();
  for (int o = blockDim.x >> 1; o > 0; o >>= 1) {
    if ((int)threadIdx.x < o) { r0[threadIdx.x] += r0[threadIdx.x + o];
                                r1[threadIdx.x] += r1[threadIdx.x + o]; }
    __syncthreads();
  }
  if (threadIdx.x == 0) { atomicAdd(&sum[c], r0[0]); atomicAdd(&sumsq[c], r1[0]); }
}

__global__ void fnet_masksum_k(const float* __restrict__ m, float* cnt, long long S) {
  float s = 0.0f;
  const long long stride = (long long)gridDim.x * blockDim.x;
  for (long long i = blockIdx.x * (long long)blockDim.x + threadIdx.x; i < S; i += stride)
    s += m[i];
  __shared__ float r0[256];
  r0[threadIdx.x] = s;
  __syncthreads();
  for (int o = blockDim.x >> 1; o > 0; o >>= 1) {
    if ((int)threadIdx.x < o) r0[threadIdx.x] += r0[threadIdx.x + o];
    __syncthreads();
  }
  if (threadIdx.x == 0) atomicAdd(cnt, r0[0]);
}

__global__ void fnet_bnapply_k(float* __restrict__ x, const float* __restrict__ m,
                               const float* __restrict__ sum, const float* __restrict__ sumsq,
                               const float* __restrict__ cnt,
                               const float* __restrict__ gamma, const float* __restrict__ beta,
                               long long S, int C)
{
  const float cn = fmaxf(cnt[0], 1.0f);
  const long long total = S * C;
  const long long stride = (long long)gridDim.x * blockDim.x;
  for (long long i = blockIdx.x * (long long)blockDim.x + threadIdx.x; i < total; i += stride) {
    const int c = (int)(i % C);
    const long long site = i / C;
    const float mean = sum[c] / cn;
    const float var  = sumsq[c] / cn - mean * mean;
    float v = (x[i] - mean) * rsqrtf(var + 1e-4f) * gamma[c] + beta[c];
    x[i] = fmaxf(v, 0.0f) * m[site];
  }
}

// scn.MaxPooling(2,3,2): max over ACTIVE sites in 3x3 window, stride 2 (VALID).
__global__ void fnet_maxpool_k(const float* __restrict__ src, const float* __restrict__ msrc,
                               float* __restrict__ dst, float* __restrict__ mdst,
                               int Bn, int H, int Wd, int C, int Ho, int Wo)
{
  const long long total = (long long)Bn * Ho * Wo * C;
  const long long stride = (long long)gridDim.x * blockDim.x;
  for (long long i = blockIdx.x * (long long)blockDim.x + threadIdx.x; i < total; i += stride) {
    const int c = (int)(i % C);
    const long long s = i / C;
    const int x = (int)(s % Wo);
    const long long t = s / Wo;
    const int y = (int)(t % Ho);
    const int b = (int)(t / Ho);
    float best = -1e30f;
    bool any = false;
#pragma unroll
    for (int ky = 0; ky < 3; ++ky) {
#pragma unroll
      for (int kx = 0; kx < 3; ++kx) {
        const int iy = y * 2 + ky, ix = x * 2 + kx;
        if (iy < H && ix < Wd) {
          const long long is = (long long)(b * H + iy) * Wd + ix;
          if (msrc[is] > 0.0f) {
            any = true;
            best = fmaxf(best, src[is * C + c]);
          }
        }
      }
    }
    dst[i] = any ? best : 0.0f;
    if (c == 0) mdst[s] = any ? 1.0f : 0.0f;
  }
}

// GRU with zero hidden state: h = mask * sigmoid(u_pre) * tanh(c_pre),
// u_pre = first C channels of the 2C-channel gate conv output.
__global__ void fnet_gru_combine_k(const float* __restrict__ g, const float* __restrict__ cb,
                                   const float* __restrict__ m, float* __restrict__ h,
                                   long long S, int C)
{
  const long long total = S * C;
  const long long stride = (long long)gridDim.x * blockDim.x;
  for (long long i = blockIdx.x * (long long)blockDim.x + threadIdx.x; i < total; i += stride) {
    const int c = (int)(i % C);
    const long long site = i / C;
    const float u = 1.0f / (1.0f + __expf(-g[site * (2 * C) + c]));
    const float cv = tanhf(cb[i]);
    h[i] = m[site] * u * cv;
  }
}

// NHWC [B,5,7,256] -> NCHW-flattened dense [B, 256*35]
__global__ void fnet_to_dense_k(const float* __restrict__ h2, float* __restrict__ dense, int Bn) {
  const long long total = (long long)Bn * 8960;
  const long long stride = (long long)gridDim.x * blockDim.x;
  for (long long i = blockIdx.x * (long long)blockDim.x + threadIdx.x; i < total; i += stride) {
    const int f = (int)(i % 8960);
    const int b = (int)(i / 8960);
    const int c = f / 35;
    const int s = f - c * 35;
    dense[i] = h2[((long long)b * 35 + s) * 256 + c];
  }
}

// ---------------------------------------------------------------------------
extern "C" void kernel_launch(void* const* d_in, const int* in_sizes, int n_in,
                              void* d_out, int out_size, void* d_ws, size_t ws_size,
                              hipStream_t stream)
{
  (void)in_sizes; (void)n_in; (void)out_size; (void)ws_size;

  const float*         x     = (const float*)d_in[0];
  const unsigned char* mask0 = (const unsigned char*)d_in[1];
  const float* g1Wg = (const float*)d_in[22];
  const float* g1Wc = (const float*)d_in[23];
  const float* g2Wg = (const float*)d_in[24];
  const float* g2Wc = (const float*)d_in[25];
  const float* W1   = (const float*)d_in[26];
  const float* b1   = (const float*)d_in[27];
  const float* W2   = (const float*)d_in[28];
  const float* b2   = (const float*)d_in[29];

  // ---- workspace carve-up (f32 elements, all blocks 16B-aligned) ----
  float* ws = (float*)d_ws;
  size_t off = 0;
  auto carve = [&](size_t n) { float* p = ws + off; off += n; return p; };
  const size_t BUFSZ = 24936960;            // 32*191*255*16 (largest map)
  float* bufA  = carve(BUFSZ);
  float* bufB  = carve(BUFSZ);
  float* m0    = carve(1558560);            // 32*191*255
  float* mlev[6];
  mlev[0] = m0;
  mlev[1] = carve(386080);                  // 32*95*127
  mlev[2] = carve(94752);                   // 32*47*63
  mlev[3] = carve(22816);                   // 32*23*31
  mlev[4] = carve(5280);                    // 32*11*15
  mlev[5] = carve(1120);                    // 32*5*7
  float* sum   = carve(256);
  float* sumsq = carve(256);
  float* cnt   = carve(4);                  // padded: keeps later carves aligned
  float* gbuf  = carve(573440);             // 1120 * 512
  float* cbuf  = carve(286720);             // 1120 * 256
  float* dense = carve(286720);             // 32 * 8960
  float* y1    = carve(32768);              // 32 * 1024

  float* outp = (float*)d_out;              // [32*5*7*12] = 13440
  float* h1   = outp + 13440;               // [32*5*7*256]
  float* h2   = h1 + 286720;

  const dim3 blk32(32), blk256(256);

  auto launch_conv = [&](const float* src, int C0, int CiW, const float* Wt,
                         const float* m, float* dst, int H, int W, int Co) {
    const long long S = 32LL * H * W;
    const int Mt = (int)((S + 15) / 16);
    if (Co == 32)
      fnet_conv3x3_wmma<2><<<dim3(Mt, 1), blk32, 0, stream>>>(src, C0, CiW, Wt, m, dst, 32, H, W, Co);
    else
      fnet_conv3x3_wmma<4><<<dim3(Mt, Co / 64), blk32, 0, stream>>>(src, C0, CiW, Wt, m, dst, 32, H, W, Co);
  };

  auto bn_relu = [&](float* buf, const float* m, const float* gamma, const float* beta,
                     long long S, int Co) {
    fnet_zero_k<<<dim3(1), blk256, 0, stream>>>(sum, 516LL);
    fnet_bnstats_k<<<dim3(Co, 32), blk256, 0, stream>>>(buf, sum, sumsq, S, Co);
    fnet_masksum_k<<<dim3(64), blk256, 0, stream>>>(m, cnt, S);
    fnet_bnapply_k<<<dim3(1024), blk256, 0, stream>>>(buf, m, sum, sumsq, cnt,
                                                      gamma, beta, S, Co);
  };

  // level-0 mask bool -> float
  fnet_mask_cvt_k<<<dim3(512), blk256, 0, stream>>>(mask0, m0, 1558560LL);

  struct Lvl { int H, W, Ci, Co, Ho, Wo; };
  const Lvl L[5] = {{191,255,  2, 16, 95,127},
                    { 95,127, 16, 32, 47, 63},
                    { 47, 63, 32, 64, 23, 31},
                    { 23, 31, 64,128, 11, 15},
                    { 11, 15,128,256,  5,  7}};

  const float* cin = x;
  float* s1 = bufA;
  float* s2 = bufB;

  for (int li = 0; li < 5; ++li) {
    const float* Wa    = (const float*)d_in[2 + 4 * li + 0];
    const float* gamma = (const float*)d_in[2 + 4 * li + 1];
    const float* beta  = (const float*)d_in[2 + 4 * li + 2];
    const float* Wb    = (const float*)d_in[2 + 4 * li + 3];
    const long long S  = 32LL * L[li].H * L[li].W;
    const int Mt = (int)((S + 15) / 16);

    // conv A: cin -> s1
    if (li == 0)
      fnet_conv3x3_smallc_wmma<1, 2><<<dim3(Mt, 1), blk32, 0, stream>>>(
          cin, Wa, mlev[li], s1, 32, L[li].H, L[li].W, 16);
    else if (li == 1)
      fnet_conv3x3_smallc_wmma<2, 16><<<dim3(Mt, 1), blk32, 0, stream>>>(
          cin, Wa, mlev[li], s1, 32, L[li].H, L[li].W, 32);
    else
      launch_conv(cin, L[li].Ci, L[li].Ci, Wa, mlev[li], s1, L[li].H, L[li].W, L[li].Co);

    // batch-norm (masked batch stats) + ReLU, in place on s1
    bn_relu(s1, mlev[li], gamma, beta, S, L[li].Co);

    // conv B: s1 -> s2
    if (li == 0)
      fnet_conv3x3_smallc_wmma<1, 16><<<dim3(Mt, 1), blk32, 0, stream>>>(
          s1, Wb, mlev[li], s2, 32, L[li].H, L[li].W, 16);
    else
      launch_conv(s1, L[li].Co, L[li].Co, Wb, mlev[li], s2, L[li].H, L[li].W, L[li].Co);

    // maxpool: s2 -> s1, mask -> mlev[li+1]
    fnet_maxpool_k<<<dim3(1024), blk256, 0, stream>>>(
        s2, mlev[li], s1, mlev[li + 1], 32, L[li].H, L[li].W, L[li].Co, L[li].Ho, L[li].Wo);

    cin = s1;
    float* t = s1; s1 = s2; s2 = t;
  }
  // cin = pooled level-5 features [1120][256], mask = mlev[5]

  const float* m5 = mlev[5];
  const long long S5 = 1120;

  // GRU 1 (h=0): gates conv (512 out, weights expect 512 in), candidate conv
  launch_conv(cin, 256, 512, g1Wg, m5, gbuf, 5, 7, 512);
  launch_conv(cin, 256, 512, g1Wc, m5, cbuf, 5, 7, 256);
  fnet_gru_combine_k<<<dim3(512), blk256, 0, stream>>>(gbuf, cbuf, m5, h1, S5, 256);

  // GRU 2 (h=0), input = h1 -> h2
  launch_conv(h1, 256, 512, g2Wg, m5, gbuf, 5, 7, 512);
  launch_conv(h1, 256, 512, g2Wc, m5, cbuf, 5, 7, 256);
  fnet_gru_combine_k<<<dim3(512), blk256, 0, stream>>>(gbuf, cbuf, m5, h2, S5, 256);

  // MLP head: dense = NCHW-flatten(h2); y1 = relu(dense @ W1 + b1); out = y1 @ W2 + b2
  fnet_to_dense_k<<<dim3(512), blk256, 0, stream>>>(h2, dense, 32);
  fnet_gemm_wmma<4><<<dim3(2, 16), blk32, 0, stream>>>(dense, W1, b1, y1, 32, 1024, 8960, 1);
  fnet_gemm_wmma<4><<<dim3(2, 7), blk32, 0, stream>>>(y1, W2, b2, outp, 32, 420, 1024, 0);
}
